// LegacyScan2DOperator_15393162788919
// MI455X (gfx1250) — matrix operand order repaired and checked by reading, not compiled, not run on hardware
//
#include <hip/hip_runtime.h>

// ---------------------------------------------------------------------------
// LegacyScan2DOperator for MI455X (gfx1250, wave32, WMMA)
//
// Pipeline (all channels-last [B,H,W,C] intermediates for coalescing):
//   1. f32->f16 converts: x, W_in, W_out
//   2. GEMM1  (WMMA f32_16x16x32_f16): proj = x @ W_in + b_in          -> buf0
//   3. depthwise 3x3 conv (SAME)                                       -> buf1
//   4. EMA scans along W (lr+rl): out = local + 0.25*(lr+rl)           -> buf0
//   5. EMA scans along H (tb+bt): out += 0.25*(tb+bt)                  -> buf0
//   6. LayerNorm over C fused with f32->f16 downconvert                -> hbuf
//   7. GEMM2  (WMMA): out = normed @ W_out + b_out                     -> d_out
//
// Workspace requirement: ~284 MB.
// ---------------------------------------------------------------------------

typedef __attribute__((ext_vector_type(16))) _Float16 v16h;
typedef __attribute__((ext_vector_type(8)))  _Float16 v8h;
typedef __attribute__((ext_vector_type(8)))  float    v8f;

union AV { v16h v; v8h h[2]; };

static constexpr int Bn = 8;
static constexpr int Hn = 96;
static constexpr int Wn = 96;
static constexpr int Dn = 192;
static constexpr int Cc = 384;            // inner dim
static constexpr int NPIX = Bn * Hn * Wn; // 73728
static constexpr float DECAY = 0.6f;
static constexpr float GAIN  = 0.4f;
static constexpr float LN_EPS = 1e-5f;

// ---------------------------------------------------------------------------
// f32 -> f16 convert (grid-stride)
// ---------------------------------------------------------------------------
__global__ void f32_to_f16_kernel(const float* __restrict__ in,
                                  _Float16* __restrict__ out, int n) {
  for (int i = blockIdx.x * blockDim.x + threadIdx.x; i < n;
       i += gridDim.x * blockDim.x) {
    out[i] = (_Float16)in[i];
  }
}

// ---------------------------------------------------------------------------
// WMMA GEMM: C[M,N] = A[M,K] (f16, row-major) * B[K,N] (f16, row-major) + bias
// One wave = one 16x16 output tile; B-fragments for the full K extent are
// hoisted and reused across `niter` M-tiles per wave. The M-tile loop is
// wave-uniform by construction: `slot` is forced into an SGPR with
// readfirstlane and the trip count is a uniform kernel arg, so the loop
// compiles to scalar branches (no EXEC masking around the WMMAs).
//
// Fragment layouts (16-bit A 16x32; B 32x16; f32 C/D): A/B use contiguous k = 16*half + e.
// ---------------------------------------------------------------------------
template <int K>
__global__ void wmma_gemm_kernel(const _Float16* __restrict__ A,
                                 const _Float16* __restrict__ Bm,
                                 const float* __restrict__ bias,
                                 float* __restrict__ Cm,
                                 int N, int slot_count, int niter) {
  constexpr int NK = K / 32;
  const int lane = threadIdx.x & 31;
  const int half = lane >> 4;   // which 16-lane half of the wave
  const int l15  = lane & 15;
  const int n0   = blockIdx.x * 16;

  // Wave-uniform slot id, forced scalar so the M loop uses SALU control flow.
  const int slot = __builtin_amdgcn_readfirstlane(
      blockIdx.y * (blockDim.x >> 5) + (threadIdx.x >> 5));

  // --- Hoist B fragments: lane holds column n0+l15, K = k0 + 16*half + e ---
  v16h bfrag[NK];
  {
    const _Float16* bcol = Bm + (n0 + l15);
#pragma unroll
    for (int kk = 0; kk < NK; ++kk) {
      const int kb = kk * 32 + 16 * half;
      v16h bv;
#pragma unroll
      for (int e = 0; e < 16; ++e) bv[e] = bcol[(size_t)(kb + e) * N];
      bfrag[kk] = bv;
    }
  }
  const float bia = bias[n0 + l15];

  for (int it = 0; it < niter; ++it) {
    const int mt = slot + it * slot_count;
    const int m0 = mt * 16;
    const _Float16* arow = A + (size_t)(m0 + l15) * K;

    // Prefetch next M-tile's A row (emits global_prefetch_b8).
    if (it + 1 < niter) {
      __builtin_prefetch(A + (size_t)((mt + slot_count) * 16 + l15) * K, 0, 0);
    }

    v8f acc = {};
#pragma unroll
    for (int kk = 0; kk < NK; ++kk) {
      const int kb = kk * 32 + 16 * half;
      AV a;
      a.h[0] = *(const v8h*)(arow + kb);        // K = kb .. kb+7
      a.h[1] = *(const v8h*)(arow + kb + 8);    // K = kb+8 .. kb+15
      acc = __builtin_amdgcn_wmma_f32_16x16x32_f16(
          false, a.v, false, bfrag[kk], (short)0, acc, false, false);
    }

    // C/D layout: VGPR r, lane -> (M = m0 + r + 8*half, N = n0 + l15)
    float* crow = Cm + (size_t)(m0 + 8 * half) * N + n0 + l15;
#pragma unroll
    for (int r = 0; r < 8; ++r) crow[(size_t)r * N] = acc[r] + bia;
  }
}

// ---------------------------------------------------------------------------
// Depthwise 3x3 conv (SAME), channels-last. One thread = one pixel x 4 chans.
// ---------------------------------------------------------------------------
__global__ void dwconv3x3_kernel(const float* __restrict__ proj,
                                 const float* __restrict__ convw,
                                 float* __restrict__ local_out) {
  const int C4 = Cc / 4;
  const int total = NPIX * C4;
  int t = blockIdx.x * blockDim.x + threadIdx.x;
  if (t >= total) return;

  const int c4 = (t % C4) * 4;
  const int p  = t / C4;
  const int w  = p % Wn;
  const int h  = (p / Wn) % Hn;
  const int b  = p / (Wn * Hn);

  // Load 4 channels x 9 taps of weights: conv_w flat [c][ky][kx] = c*9+ky*3+kx
  float wreg[4][9];
#pragma unroll
  for (int j = 0; j < 4; ++j)
#pragma unroll
    for (int k = 0; k < 9; ++k) wreg[j][k] = convw[(size_t)(c4 + j) * 9 + k];

  float a0 = 0.f, a1 = 0.f, a2 = 0.f, a3 = 0.f;
#pragma unroll
  for (int ky = -1; ky <= 1; ++ky) {
    const int hh = h + ky;
    if (hh < 0 || hh >= Hn) continue;
#pragma unroll
    for (int kx = -1; kx <= 1; ++kx) {
      const int ww = w + kx;
      if (ww < 0 || ww >= Wn) continue;
      const int tap = (ky + 1) * 3 + (kx + 1);
      const float4 v = *(const float4*)(proj +
          ((size_t)(b * Hn + hh) * Wn + ww) * Cc + c4);
      a0 += v.x * wreg[0][tap];
      a1 += v.y * wreg[1][tap];
      a2 += v.z * wreg[2][tap];
      a3 += v.w * wreg[3][tap];
    }
  }
  float4 r; r.x = a0; r.y = a1; r.z = a2; r.w = a3;
  *(float4*)(local_out + ((size_t)(b * Hn + h) * Wn + w) * Cc + c4) = r;
}

// ---------------------------------------------------------------------------
// EMA scans along W: out = local + 0.25*(lr + rl). One thread per (b,h,c);
// consecutive c -> consecutive addresses, fully coalesced per step.
// ---------------------------------------------------------------------------
__global__ void ema_w_kernel(const float* __restrict__ local,
                             float* __restrict__ out) {
  const int total = Bn * Hn * Cc;
  int t = blockIdx.x * blockDim.x + threadIdx.x;
  if (t >= total) return;
  const int c = t % Cc;
  const int h = (t / Cc) % Hn;
  const int b = t / (Cc * Hn);
  const size_t base = ((size_t)(b * Hn + h) * Wn) * Cc + c;

  float s = 0.f;
  for (int w = 0; w < Wn; ++w) {
    const size_t idx = base + (size_t)w * Cc;
    const float v = local[idx];
    s = DECAY * s + GAIN * v;
    out[idx] = v + 0.25f * s;          // local + 0.25*lr
  }
  s = 0.f;
  for (int w = Wn - 1; w >= 0; --w) {
    const size_t idx = base + (size_t)w * Cc;
    s = DECAY * s + GAIN * local[idx];
    out[idx] += 0.25f * s;             // += 0.25*rl
  }
}

// ---------------------------------------------------------------------------
// EMA scans along H: out += 0.25*(tb + bt). One thread per (b,w,c).
// ---------------------------------------------------------------------------
__global__ void ema_h_kernel(const float* __restrict__ local,
                             float* __restrict__ out) {
  const int total = Bn * Wn * Cc;
  int t = blockIdx.x * blockDim.x + threadIdx.x;
  if (t >= total) return;
  const int c = t % Cc;
  const int w = (t / Cc) % Wn;
  const int b = t / (Cc * Wn);
  const size_t base = ((size_t)b * Hn * Wn + w) * Cc + c;
  const size_t stride = (size_t)Wn * Cc;

  float s = 0.f;
  for (int h = 0; h < Hn; ++h) {
    const size_t idx = base + (size_t)h * stride;
    s = DECAY * s + GAIN * local[idx];
    out[idx] += 0.25f * s;             // += 0.25*tb
  }
  s = 0.f;
  for (int h = Hn - 1; h >= 0; --h) {
    const size_t idx = base + (size_t)h * stride;
    s = DECAY * s + GAIN * local[idx];
    out[idx] += 0.25f * s;             // += 0.25*bt
  }
}

// ---------------------------------------------------------------------------
// LayerNorm over C fused with f32->f16 downconvert. One wave32 per pixel;
// 12 elements/lane, wave reduction via shfl_xor.
// ---------------------------------------------------------------------------
__global__ void ln_to_f16_kernel(const float* __restrict__ mixed,
                                 const float* __restrict__ g,
                                 const float* __restrict__ beta,
                                 _Float16* __restrict__ out) {
  const int wave = (blockIdx.x * blockDim.x + threadIdx.x) >> 5;
  const int lane = threadIdx.x & 31;
  if (wave >= NPIX) return;
  const float* row = mixed + (size_t)wave * Cc;

  float s = 0.f, s2 = 0.f;
#pragma unroll
  for (int j = 0; j < Cc / 32; ++j) {
    const float v = row[lane + 32 * j];
    s += v;
    s2 += v * v;
  }
#pragma unroll
  for (int o = 16; o > 0; o >>= 1) {
    s  += __shfl_xor(s,  o, 32);
    s2 += __shfl_xor(s2, o, 32);
  }
  const float mu = s * (1.0f / Cc);
  const float var = s2 * (1.0f / Cc) - mu * mu;
  const float rinv = rsqrtf(var + LN_EPS);

  _Float16* orow = out + (size_t)wave * Cc;
#pragma unroll
  for (int j = 0; j < Cc / 32; ++j) {
    const int c = lane + 32 * j;
    orow[c] = (_Float16)((row[c] - mu) * rinv * g[c] + beta[c]);
  }
}

// ---------------------------------------------------------------------------
// Launch
// ---------------------------------------------------------------------------
extern "C" void kernel_launch(void* const* d_in, const int* in_sizes, int n_in,
                              void* d_out, int out_size, void* d_ws,
                              size_t ws_size, hipStream_t stream) {
  (void)in_sizes; (void)n_in; (void)out_size; (void)ws_size;

  const float* x     = (const float*)d_in[0];   // [B,H,W,D]
  const float* W_in  = (const float*)d_in[1];   // [D,C]
  const float* b_in  = (const float*)d_in[2];   // [C]
  const float* convw = (const float*)d_in[3];   // [C,1,3,3]
  const float* ln_g  = (const float*)d_in[4];   // [C]
  const float* ln_b  = (const float*)d_in[5];   // [C]
  const float* W_out = (const float*)d_in[6];   // [C,D]
  const float* b_out = (const float*)d_in[7];   // [D]
  float* out = (float*)d_out;                   // [B,H,W,D]

  // Workspace carve-up (aligned to 256B).
  char* ws = (char*)d_ws;
  size_t off = 0;
  auto carve = [&](size_t bytes) -> void* {
    void* p = ws + off;
    off = (off + bytes + 255) & ~(size_t)255;
    return p;
  };
  float*    buf0     = (float*)carve((size_t)NPIX * Cc * 4);    // proj / mixed
  float*    buf1     = (float*)carve((size_t)NPIX * Cc * 4);    // local
  _Float16* hbuf     = (_Float16*)carve((size_t)NPIX * Cc * 2); // x_f16 / normed_f16
  _Float16* w_in_h   = (_Float16*)carve((size_t)Dn * Cc * 2);
  _Float16* w_out_h  = (_Float16*)carve((size_t)Cc * Dn * 2);

  const int THREADS = 256;

  // 1) f32 -> f16 converts
  f32_to_f16_kernel<<<4096, THREADS, 0, stream>>>(x, hbuf, NPIX * Dn);
  f32_to_f16_kernel<<<144, THREADS, 0, stream>>>(W_in, w_in_h, Dn * Cc);
  f32_to_f16_kernel<<<144, THREADS, 0, stream>>>(W_out, w_out_h, Cc * Dn);

  // 2) GEMM1: proj[NPIX,C] = x_f16[NPIX,D] @ W_in_f16[D,C] + b_in
  {
    const dim3 grid(Cc / 16, 72);               // 24 N-tiles x 72 slot groups
    const int slot_count = 72 * (THREADS / 32); // 576 wave slots
    const int niter = (NPIX / 16) / slot_count; // 4608 / 576 = 8 (exact)
    wmma_gemm_kernel<Dn><<<grid, THREADS, 0, stream>>>(
        hbuf, w_in_h, b_in, buf0, Cc, slot_count, niter);
  }

  // 3) depthwise conv: buf0 -> buf1
  {
    const int total = NPIX * (Cc / 4);
    dwconv3x3_kernel<<<(total + THREADS - 1) / THREADS, THREADS, 0, stream>>>(
        buf0, convw, buf1);
  }

  // 4) EMA along W: buf1 -> buf0 (local + 0.25*(lr+rl))
  {
    const int total = Bn * Hn * Cc;
    ema_w_kernel<<<(total + THREADS - 1) / THREADS, THREADS, 0, stream>>>(
        buf1, buf0);
  }

  // 5) EMA along H: buf0 += 0.25*(tb+bt)
  {
    const int total = Bn * Wn * Cc;
    ema_h_kernel<<<(total + THREADS - 1) / THREADS, THREADS, 0, stream>>>(
        buf1, buf0);
  }

  // 6) LayerNorm + downconvert: buf0 -> hbuf (f16)
  {
    const int waves_per_block = THREADS / 32;
    const int blocks = (NPIX + waves_per_block - 1) / waves_per_block;
    ln_to_f16_kernel<<<blocks, THREADS, 0, stream>>>(buf0, ln_g, ln_b, hbuf);
  }

  // 7) GEMM2: out[NPIX,D] = normed_f16[NPIX,C] @ W_out_f16[C,D] + b_out
  {
    const dim3 grid(Dn / 16, 72);               // 12 N-tiles x 72 slot groups
    const int slot_count = 72 * (THREADS / 32); // 576 wave slots
    const int niter = (NPIX / 16) / slot_count; // 8
    wmma_gemm_kernel<Cc><<<grid, THREADS, 0, stream>>>(
        hbuf, w_out_h, b_out, out, Dn, slot_count, niter);
  }
}
